// Convolution_66511863546122
// MI455X (gfx1250) — matrix operand order repaired and checked by reading, not compiled, not run on hardware
//
#include <hip/hip_runtime.h>
#include <hip/hip_bf16.h>
#include <math.h>

// MI455X (gfx1250) fused Tensor-Field-Network convolution layer.
// One workgroup per output point 'a'; 4 waves, wave i runs radial MLP i via
// v_wmma_f32_16x16x32_f16 and accumulates its einsum over 16-row neighbor
// tiles. Software-pipelined: tile i+1's image is staged global->VGPR at the
// top of iteration i, all WMMAs run with those loads in flight, then the
// staged data drops into the ping-pong LDS buffer before a single barrier.
// W1 fragments stream from LDS (all 4 per K-slice preloaded so one dscnt wait
// covers a WMMA quad); W2/W3 fragments live in VGPRs; t0/t1 are L2-resident
// and read per-lane from global in the conv phase.

typedef _Float16 v16h __attribute__((ext_vector_type(16)));
typedef float    v8f  __attribute__((ext_vector_type(8)));

#define N_PTS 512
#define RBF   80
#define RBFP  96    // K padded to 3x32 for WMMA
#define FDIM  32
#define RU    64
#define SI    32
#define TB    16    // neighbor (b) tile rows
#define NTILE (N_PTS / TB)
#define EPS   1e-7f

__device__ __forceinline__ float sspf(float x) {
  // shifted softplus: log(0.5*exp(x)+0.5), numerically stable
  const float LN2 = 0.69314718055994531f;
  float t = (x > 20.0f) ? x : log1pf(__expf(x));
  return t - LN2;
}

__device__ __forceinline__ v8f wmma16(v16h a, v16h b, v8f c) {
  return __builtin_amdgcn_wmma_f32_16x16x32_f16(
      /*neg_a=*/false, a, /*neg_b=*/false, b,
      /*c_mod=*/(short)0, c, /*reuse_a=*/false, /*reuse_b=*/false);
}

// A fragment: 16xK f16 tile (row-major in LDS, stride ld), K-slice base k0.
// ISA layout: lane<16 -> M=lane, K = k0 + {8h+2v | v<4} and {16+8h+2(v-4)}.
__device__ __forceinline__ v16h ldsA(const _Float16* p, int ld, int lane, int k0) {
  const int m = lane & 15, h = lane >> 4;
  v16h a;
#pragma unroll
  for (int v = 0; v < 8; ++v) {
    int k = k0 + ((v < 4) ? (8 * h + 2 * v) : (16 + 8 * h + 2 * (v - 4)));
    a[2 * v]     = p[m * ld + k];
    a[2 * v + 1] = p[m * ld + k + 1];
  }
  return a;
}

// B fragment built from global f32 weights W[K][ld] (row-major), column col,
// K base k0, zero-padded past Kmax. Lane = column; K = k0 + {8h+2v | v<4} and {16+8h+2(v-4)}.
__device__ __forceinline__ v16h gblB(const float* __restrict__ W, int ld, int lane,
                                     int col, int k0, int Kmax) {
  const int h = lane >> 4;
  v16h b;
#pragma unroll
  for (int v = 0; v < 8; ++v) {
    int k = k0 + ((v < 4) ? (8 * h + 2 * v) : (16 + 8 * h + 2 * (v - 4)));
    b[2 * v]     = (k     < Kmax) ? (_Float16)W[k * ld + col]       : (_Float16)0.0f;
    b[2 * v + 1] = (k + 1 < Kmax) ? (_Float16)W[(k + 1) * ld + col] : (_Float16)0.0f;
  }
  return b;
}

__global__ __launch_bounds__(128)
void tfn_conv_kernel(const float* __restrict__ image,  const float* __restrict__ vectors,
                     const float* __restrict__ t0,     const float* __restrict__ t1,
                     const float* __restrict__ rW1,    const float* __restrict__ rb1,
                     const float* __restrict__ rW2,    const float* __restrict__ rb2,
                     const float* __restrict__ rW3,    const float* __restrict__ rb3,
                     const float* __restrict__ si_w0,  const float* __restrict__ si_w1,
                     const float* __restrict__ act_b0, const float* __restrict__ act_b1,
                     float* __restrict__ out)
{
  __shared__ __align__(32) _Float16 w1f[4][12][32][16];    // W1 B-fragments (48KB)
  __shared__ __align__(16) _Float16 imgAb[2][TB * RBFP];   // ping-pong image tile
  __shared__ _Float16 hstage[4][TB * RU];                  // per-wave restage
  __shared__ float vecsb[2][TB][4];                        // vx,vy,vz,mask per b

  // output partials overlaid on w1f (dead after the last tile's layer-1 WMMAs;
  // the final loop barrier orders the reuse)
  float* Op   = (float*)&w1f[0][0][0][0];
  float* O00  = Op;            // [32]
  float* O110 = Op + 32;       // [32]
  float* O01  = Op + 64;       // [32][3]
  float* O10  = Op + 160;      // [32][3]
  float* O111 = Op + 256;      // [32][3]

  const int a    = blockIdx.x;
  const int tid  = threadIdx.x;
  const int wave = tid >> 5;
  const int lane = tid & 31;
  const int l16  = lane & 15;
  const int half = lane >> 4;

  // ---- preload this wave's radial MLP weights as f16 WMMA B fragments ----
  const float* W1 = rW1 + wave * RBF * RU;
  const float* W2 = rW2 + wave * RU * RU;
  const float* W3 = rW3 + wave * RU * FDIM;
  v16h w2B[4][2], w3B[2][2];
#pragma unroll
  for (int nt = 0; nt < 4; ++nt) {
#pragma unroll
    for (int ks = 0; ks < 3; ++ks) {   // W1 -> LDS, fragment-native layout
      v16h b = gblB(W1, RU, lane, nt * 16 + l16, ks * 32, RBF);
      *(v16h*)&w1f[wave][nt * 3 + ks][lane][0] = b;
    }
#pragma unroll
    for (int ks = 0; ks < 2; ++ks)
      w2B[nt][ks] = gblB(W2, RU, lane, nt * 16 + l16, ks * 32, RU);
  }
#pragma unroll
  for (int nt = 0; nt < 2; ++nt)
#pragma unroll
    for (int ks = 0; ks < 2; ++ks)
      w3B[nt][ks] = gblB(W3, FDIM, lane, nt * 16 + l16, ks * 32, RU);

  float b1v[4], b2v[4], b3v[2];
#pragma unroll
  for (int nt = 0; nt < 4; ++nt) {
    b1v[nt] = rb1[wave * RU + nt * 16 + l16];
    b2v[nt] = rb2[wave * RU + nt * 16 + l16];
  }
#pragma unroll
  for (int nt = 0; nt < 2; ++nt)
    b3v[nt] = rb3[wave * FDIM + nt * 16 + l16];

  // ---- prologue: tile 0 direct to LDS; K-pad zeros for both buffers ----
  {
    const float4* img4 = (const float4*)(image + (size_t)a * N_PTS * RBF);
    for (int idx = tid; idx < TB * RBF / 4; idx += 128) {
      float4 v = img4[idx];
      int r = idx / (RBF / 4), c = (idx % (RBF / 4)) * 4;
      _Float16* dst = &imgAb[0][r * RBFP + c];
      dst[0] = (_Float16)v.x; dst[1] = (_Float16)v.y;
      dst[2] = (_Float16)v.z; dst[3] = (_Float16)v.w;
    }
    for (int idx = tid; idx < 2 * TB * 16; idx += 128) {
      int bsel = idx >= TB * 16, k = idx & (TB * 16 - 1);
      imgAb[bsel][(k >> 4) * RBFP + RBF + (k & 15)] = (_Float16)0.0f;
    }
    if (tid < TB) {
      float vx = vectors[(a * N_PTS + tid) * 3 + 0];
      float vy = vectors[(a * N_PTS + tid) * 3 + 1];
      float vz = vectors[(a * N_PTS + tid) * 3 + 2];
      float nsq = vx * vx + vy * vy + vz * vz;
      vecsb[0][tid][0] = vx; vecsb[0][tid][1] = vy; vecsb[0][tid][2] = vz;
      vecsb[0][tid][3] = (nsq < EPS * EPS) ? 0.0f : 1.0f;        // mask_radial
    }
  }
  __syncthreads();

  // per-lane einsum accumulators; lane owns f = 16*t + l16 for t in {0,1}
  float a00[2]    = {0.f, 0.f};
  float a110[2]   = {0.f, 0.f};
  float a01[3][2] = {{0.f,0.f},{0.f,0.f},{0.f,0.f}};
  float a10[3][2] = {{0.f,0.f},{0.f,0.f},{0.f,0.f}};
  float a111[3][2]= {{0.f,0.f},{0.f,0.f},{0.f,0.f}};

  for (int it = 0; it < NTILE; ++it) {
    const int cur = it & 1, nxt = cur ^ 1;
    const int b0  = it * TB;
    const int bn  = b0 + TB;                 // next tile base
    const bool more = (bn < N_PTS);

    // ---- stage tile it+1's image: issue global loads now, consume later ----
    float4 irv0, irv1, irv2;
    float  vvx = 0.f, vvy = 0.f, vvz = 0.f;
    if (more) {
      const float4* img4 = (const float4*)(image + (size_t)(a * N_PTS + bn) * RBF);
      irv0 = img4[tid];
      irv1 = img4[tid + 128];
      if (tid < 64) irv2 = img4[tid + 256];
      if (tid < TB) {
        vvx = vectors[(a * N_PTS + bn + tid) * 3 + 0];
        vvy = vectors[(a * N_PTS + bn + tid) * 3 + 1];
        vvz = vectors[(a * N_PTS + bn + tid) * 3 + 2];
      }
      if (bn + TB < N_PTS && tid < 40) {     // warm L2 two tiles ahead
        const char* p2 = (const char*)(image + (size_t)(a * N_PTS + bn + TB) * RBF);
        __builtin_prefetch(p2 + tid * 128, 0, 1);
      }
    }

    // ---- radial MLP: layer 1 (16x96 @ 96x64), W1 fragments from LDS ----
    // All 4 fragments of a K-slice are loaded into distinct registers before
    // the WMMA quad so a single s_wait_dscnt covers them and the DS pipe
    // stays ahead of the matrix pipe.
    const _Float16* imgA = imgAb[cur];
    v8f h1[4];
#pragma unroll
    for (int nt = 0; nt < 4; ++nt) h1[nt] = (v8f)0.0f;
#pragma unroll
    for (int ks = 0; ks < 3; ++ks) {
      v16h A  = ldsA(imgA, RBFP, lane, ks * 32);
      v16h B0 = *(const v16h*)&w1f[wave][0 * 3 + ks][lane][0];
      v16h B1 = *(const v16h*)&w1f[wave][1 * 3 + ks][lane][0];
      v16h B2 = *(const v16h*)&w1f[wave][2 * 3 + ks][lane][0];
      v16h B3 = *(const v16h*)&w1f[wave][3 * 3 + ks][lane][0];
      h1[0] = wmma16(A, B0, h1[0]);
      h1[1] = wmma16(A, B1, h1[1]);
      h1[2] = wmma16(A, B2, h1[2]);
      h1[3] = wmma16(A, B3, h1[3]);
    }
    _Float16* hs = hstage[wave];
#pragma unroll
    for (int nt = 0; nt < 4; ++nt)
#pragma unroll
      for (int j = 0; j < 8; ++j)                     // D layout -> row-major
        hs[(j + 8 * half) * RU + nt * 16 + l16] = (_Float16)(h1[nt][j] + b1v[nt]);

    // ---- layer 2 (16x64 @ 64x64), W2 fragments in VGPRs ----
    v8f h2[4];
#pragma unroll
    for (int nt = 0; nt < 4; ++nt) h2[nt] = (v8f)0.0f;
#pragma unroll
    for (int ks = 0; ks < 2; ++ks) {
      v16h A = ldsA(hs, RU, lane, ks * 32);
#pragma unroll
      for (int nt = 0; nt < 4; ++nt) h2[nt] = wmma16(A, w2B[nt][ks], h2[nt]);
    }
#pragma unroll
    for (int nt = 0; nt < 4; ++nt)
#pragma unroll
      for (int j = 0; j < 8; ++j)
        hs[(j + 8 * half) * RU + nt * 16 + l16] = (_Float16)(h2[nt][j] + b2v[nt]);

    // ---- layer 3 (16x64 @ 64x32) -> radial tile R[b,f] in D layout ----
    v8f rr[2];
    rr[0] = (v8f)0.0f; rr[1] = (v8f)0.0f;
#pragma unroll
    for (int ks = 0; ks < 2; ++ks) {
      v16h A = ldsA(hs, RU, lane, ks * 32);
#pragma unroll
      for (int nt = 0; nt < 2; ++nt) rr[nt] = wmma16(A, w3B[nt][ks], rr[nt]);
    }

    // ---- point-convolution accumulation; t0/t1 are L2-resident, read
    //      per-lane straight from global (b32 / b96) ----
#pragma unroll
    for (int t = 0; t < 2; ++t) {
      const int f = t * 16 + l16;
#pragma unroll
      for (int j = 0; j < 8; ++j) {
        const int b  = j + 8 * half;
        const int bg = b0 + b;
        const float rv = rr[t][j] + b3v[t];
        if (wave == 0) {                               // out00
          a00[t] += rv * t0[bg * FDIM + f];
        } else if (wave == 1) {                        // out01 (filt1 of radial 1)
          float s = rv * vecsb[cur][b][3] * t0[bg * FDIM + f];
          a01[0][t] += s * vecsb[cur][b][0];
          a01[1][t] += s * vecsb[cur][b][1];
          a01[2][t] += s * vecsb[cur][b][2];
        } else if (wave == 2) {                        // out10
          const float* tp = t1 + (size_t)(bg * FDIM + f) * 3;
          a10[0][t] += rv * tp[0];
          a10[1][t] += rv * tp[1];
          a10[2][t] += rv * tp[2];
        } else {                                       // out110 + out111 (cross)
          const float* tp = t1 + (size_t)(bg * FDIM + f) * 3;
          float m  = rv * vecsb[cur][b][3];
          float vx = vecsb[cur][b][0], vy = vecsb[cur][b][1], vz = vecsb[cur][b][2];
          float tx = tp[0], ty = tp[1], tz = tp[2];
          a110[t]    += m * (vx * tx + vy * ty + vz * tz);
          a111[0][t] += m * (vy * tz - vz * ty);
          a111[1][t] += m * (vz * tx - vx * tz);
          a111[2][t] += m * (vx * ty - vy * tx);
        }
      }
    }

    // ---- drop staged tile it+1 into the other LDS buffer ----
    if (more) {
      {
        int idx = tid, r = idx / (RBF / 4), c = (idx % (RBF / 4)) * 4;
        _Float16* d = &imgAb[nxt][r * RBFP + c];
        d[0] = (_Float16)irv0.x; d[1] = (_Float16)irv0.y;
        d[2] = (_Float16)irv0.z; d[3] = (_Float16)irv0.w;
      }
      {
        int idx = tid + 128, r = idx / (RBF / 4), c = (idx % (RBF / 4)) * 4;
        _Float16* d = &imgAb[nxt][r * RBFP + c];
        d[0] = (_Float16)irv1.x; d[1] = (_Float16)irv1.y;
        d[2] = (_Float16)irv1.z; d[3] = (_Float16)irv1.w;
      }
      if (tid < 64) {
        int idx = tid + 256, r = idx / (RBF / 4), c = (idx % (RBF / 4)) * 4;
        _Float16* d = &imgAb[nxt][r * RBFP + c];
        d[0] = (_Float16)irv2.x; d[1] = (_Float16)irv2.y;
        d[2] = (_Float16)irv2.z; d[3] = (_Float16)irv2.w;
      }
      if (tid < TB) {
        float nsq = vvx * vvx + vvy * vvy + vvz * vvz;
        vecsb[nxt][tid][0] = vvx; vecsb[nxt][tid][1] = vvy; vecsb[nxt][tid][2] = vvz;
        vecsb[nxt][tid][3] = (nsq < EPS * EPS) ? 0.0f : 1.0f;
      }
    }
    __syncthreads();   // single barrier: tile it consumed, tile it+1 published
  }

  // ---- fold the two lane halves (b-split) and publish per-f partials ----
  // (O* arrays overlay w1f, which is dead now; the loop's final barrier ordered it)
#pragma unroll
  for (int t = 0; t < 2; ++t) {
    const int f = t * 16 + l16;
    if (wave == 0) {
      float s = a00[t] + __shfl_xor(a00[t], 16, 32);
      if (half == 0) O00[f] = s;
    } else if (wave == 1) {
#pragma unroll
      for (int i = 0; i < 3; ++i) {
        float s = a01[i][t] + __shfl_xor(a01[i][t], 16, 32);
        if (half == 0) O01[f * 3 + i] = s;
      }
    } else if (wave == 2) {
#pragma unroll
      for (int i = 0; i < 3; ++i) {
        float s = a10[i][t] + __shfl_xor(a10[i][t], 16, 32);
        if (half == 0) O10[f * 3 + i] = s;
      }
    } else {
      float s = a110[t] + __shfl_xor(a110[t], 16, 32);
      if (half == 0) O110[f] = s;
#pragma unroll
      for (int i = 0; i < 3; ++i) {
        float s1 = a111[i][t] + __shfl_xor(a111[i][t], 16, 32);
        if (half == 0) O111[f * 3 + i] = s1;
      }
    }
  }
  __syncthreads();

  // ---- self interaction + equivariant shifted-softplus activation ----
  if (tid < SI) {
    const int g = tid;
    float r0 = 0.0f;
    for (int f = 0; f < FDIM; ++f) {
      r0 += O00[f]  * si_w0[g * (2 * FDIM) + f];
      r0 += O110[f] * si_w0[g * (2 * FDIM) + FDIM + f];
    }
    float r1[3] = {0.f, 0.f, 0.f};
    for (int f = 0; f < FDIM; ++f) {
      float w0w = si_w1[g * (3 * FDIM) + f];
      float w1w = si_w1[g * (3 * FDIM) + FDIM + f];
      float w2w = si_w1[g * (3 * FDIM) + 2 * FDIM + f];
#pragma unroll
      for (int i = 0; i < 3; ++i)
        r1[i] += O01[f * 3 + i] * w0w + O10[f * 3 + i] * w1w + O111[f * 3 + i] * w2w;
    }
    out[a * SI + g] = sspf(r0 + act_b0[g]);                 // y0: (N,SI,1)
    float nsq = r1[0] * r1[0] + r1[1] * r1[1] + r1[2] * r1[2];
    float n1  = sqrtf(fmaxf(nsq, EPS));
    float sc  = sspf(n1 + act_b1[g]) / n1;
#pragma unroll
    for (int i = 0; i < 3; ++i)                             // y1: (N,SI,3)
      out[N_PTS * SI + (a * SI + g) * 3 + i] = r1[i] * sc;
  }
}

extern "C" void kernel_launch(void* const* d_in, const int* in_sizes, int n_in,
                              void* d_out, int out_size, void* d_ws, size_t ws_size,
                              hipStream_t stream) {
  (void)in_sizes; (void)n_in; (void)out_size; (void)d_ws; (void)ws_size;
  const float* image  = (const float*)d_in[0];
  const float* vecs   = (const float*)d_in[1];
  const float* t0     = (const float*)d_in[2];
  const float* t1     = (const float*)d_in[3];
  const float* rW1    = (const float*)d_in[4];
  const float* rb1    = (const float*)d_in[5];
  const float* rW2    = (const float*)d_in[6];
  const float* rb2    = (const float*)d_in[7];
  const float* rW3    = (const float*)d_in[8];
  const float* rb3    = (const float*)d_in[9];
  const float* si_w0  = (const float*)d_in[10];
  const float* si_w1  = (const float*)d_in[11];
  const float* act_b0 = (const float*)d_in[12];
  const float* act_b1 = (const float*)d_in[13];
  float* out = (float*)d_out;

  tfn_conv_kernel<<<N_PTS, 128, 0, stream>>>(image, vecs, t0, t1,
                                             rW1, rb1, rW2, rb2, rW3, rb3,
                                             si_w0, si_w1, act_b0, act_b1, out);
}
